// VAE_65300682768867
// MI455X (gfx1250) — compile-verified, matrix operand order checked by reading, not run on hardware
//
#include <hip/hip_runtime.h>
#include <hip/hip_bf16.h>
#include <math.h>

typedef __attribute__((ext_vector_type(16))) _Float16 v16h;
typedef __attribute__((ext_vector_type(8)))  _Float16 v8h;
typedef __attribute__((ext_vector_type(8)))  float    v8f;
typedef int i32x4 __attribute__((vector_size(16)));   // matches async builtin pointee

#define ACT_NONE 0
#define ACT_ELU  1
#define ACT_TANH 2

// ---------------------------------------------------------------------------
// Tiled GEMM: C[M,N] = act(A[M,K] @ B[K,N] + bias[N])
// Block tile 128x64, 8 waves (wave32), each wave computes a 32x32 tile via
// 2x2 v_wmma_f32_16x16x32_f16 with f32 accumulation.
// A tile: f32 staged via async global->LDS DMA (gfx1250), converted to f16
//         at fragment build (v_cvt_pk_f16_f32 co-issues with WMMA).
// B tile: transposed + converted to f16 once at staging; fragments are two
//         aligned ds_load_b128 per v16h, no per-use conversion.
// ---------------------------------------------------------------------------
template <int ACT>
__launch_bounds__(256)
__global__ void gemm_wmma_f16(const float* __restrict__ A,
                              const float* __restrict__ B,
                              const float* __restrict__ bias,
                              float* __restrict__ C,
                              int M, int K, int N)
{
    constexpr int BM = 128, BN = 64, BK = 32;
    constexpr int LDA  = BK + 4;   // f32 dwords: 36 -> 144B rows, 16B aligned
    constexpr int LDBH = BK + 8;   // f16 halfwords: 40 -> 80B rows, 16B aligned

    __shared__ float    As[BM][LDA];    // f32 staging, row-major [m][k]
    __shared__ _Float16 Bs[BN][LDBH];   // f16 staging, TRANSPOSED [n][k]

    const int tid    = threadIdx.x;
    const int wave   = tid >> 5;        // 0..7
    const int lane   = tid & 31;
    const int wm     = wave >> 1;       // 0..3 (M direction)
    const int wn     = wave & 1;        // 0..1 (N direction)
    const int lane16 = lane & 15;
    const int lgrp   = lane >> 4;       // lane half (0/1)

    const int bm = blockIdx.y * BM;
    const int bn = blockIdx.x * BN;

    v8f acc[2][2] = {};

    for (int kt = 0; kt < K; kt += BK) {
        // ---------------- A tile: 128x32 f32 ----------------
        {
            const int r = tid >> 3;            // 0..31
            const int c = (tid & 7) * 4;       // 0..28 step 4
            const bool full_k = (kt + BK <= K);
#if __has_builtin(__builtin_amdgcn_global_load_async_to_lds_b128)
            if (full_k) {
                #pragma unroll
                for (int rr = 0; rr < 4; ++rr) {
                    const int row = r + rr * 32;
                    const float* src = A + (size_t)(bm + row) * K + kt + c;
                    __builtin_amdgcn_global_load_async_to_lds_b128(
                        (__attribute__((address_space(1))) i32x4*)src,
                        (__attribute__((address_space(3))) i32x4*)&As[row][c],
                        0, 0);
                }
            } else
#endif
            {
                #pragma unroll
                for (int rr = 0; rr < 4; ++rr) {
                    const int row = r + rr * 32;
                    const float* src = A + (size_t)(bm + row) * K + kt + c;
                    float4 t;
                    if (full_k || (kt + c + 3 < K)) {
                        t = *(const float4*)src;
                    } else {
                        t.x = (kt + c + 0 < K) ? src[0] : 0.0f;
                        t.y = (kt + c + 1 < K) ? src[1] : 0.0f;
                        t.z = (kt + c + 2 < K) ? src[2] : 0.0f;
                        t.w = (kt + c + 3 < K) ? src[3] : 0.0f;
                    }
                    As[row][c + 0] = t.x; As[row][c + 1] = t.y;
                    As[row][c + 2] = t.z; As[row][c + 3] = t.w;
                }
            }
            // software prefetch of the next K tile (global_prefetch_b8)
            if (kt + BK < K)
                __builtin_prefetch(A + (size_t)(bm + r) * K + kt + BK + c, 0, 1);
        }

        // ---- B tile: 32x64 f32 -> transpose + convert -> f16 LDS [n][k] ----
        {
            const int kr = tid >> 3;           // 0..31 (k within tile)
            const int c0 = (tid & 7) * 4;
            const bool krow_ok = (kt + kr) < K;
            #pragma unroll
            for (int cc = 0; cc < 2; ++cc) {
                const int c = c0 + cc * 32;    // 0..63
                const float* src = B + (size_t)(kt + kr) * N + bn + c;
                #pragma unroll
                for (int u = 0; u < 4; ++u) {
                    const float vv =
                        (krow_ok && (bn + c + u) < N) ? src[u] : 0.0f;
                    Bs[c + u][kr] = (_Float16)vv;
                }
            }
            if (kt + BK < K)
                __builtin_prefetch(B + (size_t)(kt + BK + kr) * N + bn + c0, 0, 1);
        }

#if __has_builtin(__builtin_amdgcn_global_load_async_to_lds_b128)
#if __has_builtin(__builtin_amdgcn_s_wait_asynccnt)
        __builtin_amdgcn_s_wait_asynccnt(0);
#else
        asm volatile("s_wait_asynccnt 0" ::: "memory");
#endif
#endif
        __syncthreads();

        // ------------- fragments + WMMA -------------
        // ISA 16-bit operand layout: lane element e <-> k = (e&7)+(e>>3)*16+lgrp*8
        // => two contiguous 8-halfword runs at k = lgrp*8 and k = 16+lgrp*8.
        v16h bfrag[2];
        #pragma unroll
        for (int tn = 0; tn < 2; ++tn) {
            const int ncol = wn * 32 + tn * 16 + lane16;
            const v8h blo = *(const v8h*)&Bs[ncol][lgrp * 8];
            const v8h bhi = *(const v8h*)&Bs[ncol][16 + lgrp * 8];
            bfrag[tn] = __builtin_shufflevector(blo, bhi,
                0, 1, 2, 3, 4, 5, 6, 7, 8, 9, 10, 11, 12, 13, 14, 15);
        }
        #pragma unroll
        for (int tm = 0; tm < 2; ++tm) {
            const int mrow = wm * 32 + tm * 16 + lane16;
            v16h afrag;
            #pragma unroll
            for (int e = 0; e < 16; ++e) {
                const int k = (e & 7) + ((e >> 3) << 4) + lgrp * 8;
                afrag[e] = (_Float16)As[mrow][k];
            }
            #pragma unroll
            for (int tn = 0; tn < 2; ++tn) {
                acc[tm][tn] = __builtin_amdgcn_wmma_f32_16x16x32_f16(
                    false, afrag, false, bfrag[tn],
                    (short)0, acc[tm][tn], false, false);
            }
        }
        __syncthreads();
    }

    // ------------------------- epilogue -------------------------
    // C/D layout: VGPR v, lanes 0-15 -> M=v, lanes 16-31 -> M=v+8, N=lane&15
    #pragma unroll
    for (int tm = 0; tm < 2; ++tm) {
        #pragma unroll
        for (int tn = 0; tn < 2; ++tn) {
            const int ncol = bn + wn * 32 + tn * 16 + lane16;
            if (ncol < N) {
                const float bv = bias[ncol];
                #pragma unroll
                for (int v = 0; v < 8; ++v) {
                    const int mrow = bm + wm * 32 + tm * 16 + lgrp * 8 + v;
                    if (mrow < M) {
                        float x = acc[tm][tn][v] + bv;
                        if (ACT == ACT_ELU)       x = (x > 0.0f) ? x : (expf(x) - 1.0f);
                        else if (ACT == ACT_TANH) x = tanhf(x);
                        C[(size_t)mrow * N + ncol] = x;
                    }
                }
            }
        }
    }
}

// ---------------------------------------------------------------------------
// Latent kernel: Bmat = W * relu(j-i); eps = exp(logvar/2)*noise;
// forward substitution z (I - B)^T-solve; Bz = z @ Bmat.
// One thread per batch row; Bmat cached in LDS.
// ---------------------------------------------------------------------------
__launch_bounds__(256)
__global__ void latent_kernel(const float* __restrict__ logvar,
                              const float* __restrict__ eps_noise,
                              const float* __restrict__ W,
                              float* __restrict__ z_out,
                              float* __restrict__ bz_out,
                              float* __restrict__ bmat_out)
{
    __shared__ float Bsm[16][16];
    const int t = threadIdx.x;
    {
        const int i = t >> 4, j = t & 15;
        const float v = W[i * 16 + j] * fmaxf((float)(j - i), 0.0f);
        Bsm[i][j] = v;
        if (blockIdx.x == 0) bmat_out[i * 16 + j] = v;   // strictly upper tri
    }
    __syncthreads();

    const int row = blockIdx.x * 256 + t;   // 0..1023
    float z[16];
    #pragma unroll
    for (int j = 0; j < 16; ++j) {
        float s = expf(0.5f * logvar[row * 16 + j]) * eps_noise[row * 16 + j];
        #pragma unroll
        for (int i = 0; i < 16; ++i)
            if (i < j) s += z[i] * Bsm[i][j];
        z[j] = s;
    }
    #pragma unroll
    for (int j = 0; j < 16; ++j) {
        z_out[row * 16 + j] = z[j];
        float s = 0.0f;
        #pragma unroll
        for (int i = 0; i < 16; ++i)
            if (i < j) s += z[i] * Bsm[i][j];
        bz_out[row * 16 + j] = s;
    }
}

// ---------------------------------------------------------------------------
extern "C" void kernel_launch(void* const* d_in, const int* in_sizes, int n_in,
                              void* d_out, int out_size, void* d_ws, size_t ws_size,
                              hipStream_t stream)
{
    (void)in_sizes; (void)n_in; (void)out_size; (void)ws_size;

    const float* x        = (const float*)d_in[0];   // [1024, 27648]
    const float* eps      = (const float*)d_in[1];   // [1024, 16]
    const float* enc_w1   = (const float*)d_in[2];   // [27648, 900]
    const float* enc_b1   = (const float*)d_in[3];
    const float* enc_w2   = (const float*)d_in[4];   // [900, 300]
    const float* enc_b2   = (const float*)d_in[5];
    const float* logvar_w = (const float*)d_in[6];   // [300, 16]
    const float* logvar_b = (const float*)d_in[7];
    const float* W        = (const float*)d_in[8];   // [16, 16]
    const float* dec_w1   = (const float*)d_in[9];   // [16, 300]
    const float* dec_b1   = (const float*)d_in[10];
    const float* dec_w2   = (const float*)d_in[11];  // [300, 300]
    const float* dec_b2   = (const float*)d_in[12];
    const float* dec_w3   = (const float*)d_in[13];  // [300, 27648]
    const float* dec_b3   = (const float*)d_in[14];

    float* out        = (float*)d_out;
    float* z_out      = out;                 // [1024,16]
    float* logvar_out = out + 16384;         // [1024,16]
    float* bz_out     = out + 32768;         // [1024,16]
    float* bmat_out   = out + 49152;         // [16,16]
    float* xhat       = out + 49408;         // [1024,27648]

    float* h1 = (float*)d_ws;                // [1024, 900]
    float* h2 = h1 + (size_t)1024 * 900;     // [1024, 300]
    float* g1 = h2 + (size_t)1024 * 300;     // [1024, 300]
    float* g2 = g1 + (size_t)1024 * 300;     // [1024, 300]

    const dim3 blk(256);

    // encoder
    gemm_wmma_f16<ACT_ELU ><<<dim3(15, 8), blk, 0, stream>>>(x,  enc_w1, enc_b1, h1, 1024, 27648, 900);
    gemm_wmma_f16<ACT_ELU ><<<dim3( 5, 8), blk, 0, stream>>>(h1, enc_w2, enc_b2, h2, 1024,   900, 300);
    gemm_wmma_f16<ACT_NONE><<<dim3( 1, 8), blk, 0, stream>>>(h2, logvar_w, logvar_b, logvar_out, 1024, 300, 16);

    // latent triangular solve + Bz + Bmat
    latent_kernel<<<dim3(4), blk, 0, stream>>>(logvar_out, eps, W, z_out, bz_out, bmat_out);

    // decoder
    gemm_wmma_f16<ACT_ELU ><<<dim3(  5, 8), blk, 0, stream>>>(z_out, dec_w1, dec_b1, g1, 1024,  16, 300);
    gemm_wmma_f16<ACT_ELU ><<<dim3(  5, 8), blk, 0, stream>>>(g1,    dec_w2, dec_b2, g2, 1024, 300, 300);
    gemm_wmma_f16<ACT_TANH><<<dim3(432, 8), blk, 0, stream>>>(g2,    dec_w3, dec_b3, xhat, 1024, 300, 27648);
}
